// GraphNet_16569983828526
// MI455X (gfx1250) — compile-verified, matrix operand order checked by reading, not run on hardware
//
#include <hip/hip_runtime.h>
#include <hip/hip_bf16.h>

typedef __attribute__((ext_vector_type(16))) __bf16 v16bf;
typedef __attribute__((ext_vector_type(8)))  float  v8f;

#define B_  64
#define N_  8192
#define E_  65536
#define G_  32
#define K_  70
#define KP_ 80     // K padded to multiple of 16 (5 M-tiles)
#define SPLIT 32   // N-slices per graph in pooling kernel (2048 waves in flight)

// ---------------- degree / normalization ----------------
__global__ void k_init_deg(float* dinv) {
    int i = blockIdx.x * blockDim.x + threadIdx.x;
    if (i < B_ * N_) dinv[i] = 1.0f;   // self loop
}

__global__ void k_count_deg(const int* __restrict__ ei, float* dinv) {
    int i = blockIdx.x * blockDim.x + threadIdx.x;
    if (i >= B_ * E_) return;
    int b = i / E_, e = i - b * E_;
    int d = ei[b * 2 * E_ + E_ + e];
    atomicAdd(&dinv[b * N_ + d], 1.0f);
}

__global__ void k_rsqrt_deg(float* dinv) {
    int i = blockIdx.x * blockDim.x + threadIdx.x;
    if (i < B_ * N_) dinv[i] = rsqrtf(dinv[i]);
}

// ---------------- GCN layer pieces ----------------
template<int IN, int OUT>
__global__ void k_linear(const float* __restrict__ x, const float* __restrict__ W,
                         float* __restrict__ h) {
    int i = blockIdx.x * blockDim.x + threadIdx.x;
    if (i >= B_ * N_) return;
    const float* xr = x + (long long)i * IN;
    float* hr = h + (long long)i * OUT;
    float xi[IN];
#pragma unroll
    for (int k = 0; k < IN; k++) xi[k] = xr[k];
#pragma unroll
    for (int o = 0; o < OUT; o++) {
        float acc = 0.0f;
#pragma unroll
        for (int k = 0; k < IN; k++) acc += xi[k] * W[k * OUT + o];
        hr[o] = acc;
    }
}

// out = h * dinv^2 (self-loop term, also initializes accumulation target)
template<int OUT>
__global__ void k_selfinit(const float* __restrict__ h, const float* __restrict__ dinv,
                           float* __restrict__ out) {
    int i = blockIdx.x * blockDim.x + threadIdx.x;
    if (i >= B_ * N_) return;
    float di = dinv[i];
    float nm = di * di;
    const float* hr = h + (long long)i * OUT;
    float* orow = out + (long long)i * OUT;
#pragma unroll
    for (int o = 0; o < OUT; o++) orow[o] = hr[o] * nm;
}

template<int OUT>
__global__ void k_scatter(const float* __restrict__ h, const float* __restrict__ dinv,
                          const int* __restrict__ ei, float* out) {
    int i = blockIdx.x * blockDim.x + threadIdx.x;
    if (i >= B_ * E_) return;
    int b = i / E_, e = i - b * E_;
    int s = ei[b * 2 * E_ + e];
    int d = ei[b * 2 * E_ + E_ + e];
    float nm = dinv[b * N_ + s] * dinv[b * N_ + d];
    const float* hs = h + ((long long)b * N_ + s) * OUT;
    float* od = out + ((long long)b * N_ + d) * OUT;
#pragma unroll
    for (int o = 0; o < OUT; o++) atomicAdd(&od[o], hs[o] * nm);
}

template<int OUT>
__global__ void k_biasrelu(float* x, const float* __restrict__ bias) {
    int i = blockIdx.x * blockDim.x + threadIdx.x;
    if (i >= B_ * N_ * OUT) return;
    int o = i % OUT;
    float v = x[i] + bias[o];
    x[i] = v > 0.0f ? v : 0.0f;
}

__global__ void k_zero_out(float* out, int n) {
    int i = blockIdx.x * blockDim.x + threadIdx.x;
    if (i < n) out[i] = 0.0f;
}

// ---------------- fused gumbel-softmax + WMMA pooling ----------------
// One wave (32 threads) per (graph, N-slice). pooled[k,g] += sum_n p[n,k]*x3[n,g]
__global__ __launch_bounds__(32)
void k_pool(const float* __restrict__ x3, const float* __restrict__ logits,
            const float* __restrict__ gu, float* out) {
    __shared__ float ls_s[32][KP_];  // per-chunk sampled probs, padded K
    __shared__ float ls_x[32][G_];   // per-chunk node features

    int b    = blockIdx.x / SPLIT;
    int part = blockIdx.x % SPLIT;
    int lane = threadIdx.x;
    int m    = lane & 15;
    int half = lane >> 4;
    const int nodesPer = N_ / SPLIT;   // 256
    int n0 = part * nodesPer;

    v8f acc[5][2];
#pragma unroll
    for (int mt = 0; mt < 5; mt++)
#pragma unroll
        for (int nt = 0; nt < 2; nt++)
#pragma unroll
            for (int r = 0; r < 8; r++) acc[mt][nt][r] = 0.0f;

    for (int c = 0; c < nodesPer; c += 32) {
        long long nbase = (long long)b * N_ + n0 + c;
        const float* lr = logits + nbase * K_;   // 32*70 contiguous floats
        const float* ur = gu     + nbase * K_;
        const float* xr = x3     + nbase * G_;   // 32*32 contiguous floats

        // ---- cooperative, fully-coalesced staging; gumbel fused in ----
        {
            int k = lane, node = 0;              // idx = lane + 32*s ; k = idx%70, node = idx/70
#pragma unroll 4
            for (int s = 0; s < (32 * K_) / 32; s++) {
                int idx = lane + 32 * s;
                float u = ur[idx];
                u = fminf(fmaxf(u, 1e-6f), 1.0f - 1e-6f);
                float gmb = -__logf(-__logf(u));
                ls_s[node][k] = (lr[idx] + gmb) * 2.0f;   // /TEMP, TEMP=0.5
                k += 32;
                if (k >= K_) { k -= K_; node += 1; }      // 32 < 70: one wrap max
            }
        }
#pragma unroll
        for (int s = 0; s < (32 * (KP_ - K_)) / 32; s++) {  // zero pad columns 70..79
            int idx = lane + 32 * s;
            ls_s[idx / (KP_ - K_)][K_ + idx % (KP_ - K_)] = 0.0f;
        }
#pragma unroll
        for (int s = 0; s < (32 * G_) / 32; s++)            // features, coalesced
            ls_x[s][lane] = xr[s * G_ + lane];
        __syncthreads();

        // prefetch next chunk's dominant stream (global_prefetch_b8)
        if (c + 32 < nodesPer) {
            __builtin_prefetch(lr + 32 * K_ + lane * K_, 0, 0);
            __builtin_prefetch(ur + 32 * K_ + lane * K_, 0, 0);
        }

        // ---- per-node softmax out of LDS ----
        float mx = -1e30f;
#pragma unroll 5
        for (int k = 0; k < K_; k++) mx = fmaxf(mx, ls_s[lane][k]);
        float sum = 0.0f;
#pragma unroll 5
        for (int k = 0; k < K_; k++) {
            float e = __expf(ls_s[lane][k] - mx);
            ls_s[lane][k] = e;
            sum += e;
        }
        float inv = 1.0f / sum;
#pragma unroll 5
        for (int k = 0; k < K_; k++) ls_s[lane][k] *= inv;
        __syncthreads();

        // ---- fragments + WMMA (ISA 16-bit A/B lane layouts) ----
        v16bf bf[2];
#pragma unroll
        for (int nt = 0; nt < 2; nt++) {
#pragma unroll
            for (int j = 0; j < 8; j++) {
                int kk = (j < 4 ? 2 * j : 16 + 2 * (j - 4)) + 8 * half;
                bf[nt][2 * j]     = (__bf16)ls_x[kk][nt * 16 + m];
                bf[nt][2 * j + 1] = (__bf16)ls_x[kk + 1][nt * 16 + m];
            }
        }
#pragma unroll
        for (int mt = 0; mt < 5; mt++) {
            v16bf af;
#pragma unroll
            for (int j = 0; j < 8; j++) {
                int kk = (j < 4 ? 2 * j : 16 + 2 * (j - 4)) + 8 * half;
                af[2 * j]     = (__bf16)ls_s[kk][mt * 16 + m];
                af[2 * j + 1] = (__bf16)ls_s[kk + 1][mt * 16 + m];
            }
#pragma unroll
            for (int nt = 0; nt < 2; nt++) {
                acc[mt][nt] = __builtin_amdgcn_wmma_f32_16x16x32_bf16(
                    false, af, false, bf[nt], (short)0, acc[mt][nt], false, false);
            }
        }
        __syncthreads();
    }

    // C/D layout: VGPR r, lanes 0-15 -> M=r, lanes 16-31 -> M=r+8; N = lane&15
#pragma unroll
    for (int mt = 0; mt < 5; mt++)
#pragma unroll
        for (int nt = 0; nt < 2; nt++)
#pragma unroll
            for (int r = 0; r < 8; r++) {
                int cat  = mt * 16 + r + 8 * half;
                int feat = nt * 16 + m;
                if (cat < K_)
                    atomicAdd(&out[((long long)b * K_ + cat) * G_ + feat], acc[mt][nt][r]);
            }
}

// ---------------- host launch ----------------
extern "C" void kernel_launch(void* const* d_in, const int* in_sizes, int n_in,
                              void* d_out, int out_size, void* d_ws, size_t ws_size,
                              hipStream_t stream) {
    const float* nodes  = (const float*)d_in[0];
    const int*   ei     = (const int*)  d_in[1];
    const float* W1     = (const float*)d_in[2];
    const float* b1     = (const float*)d_in[3];
    const float* W2     = (const float*)d_in[4];
    const float* b2     = (const float*)d_in[5];
    const float* W3     = (const float*)d_in[6];
    const float* b3     = (const float*)d_in[7];
    const float* logits = (const float*)d_in[8];
    const float* gu     = (const float*)d_in[9];
    float* out = (float*)d_out;

    float* ws    = (float*)d_ws;
    float* dinv  = ws;                                // B*N
    float* buf_h = dinv  + (size_t)B_ * N_;           // B*N*32
    float* buf_x = buf_h + (size_t)B_ * N_ * 32;      // B*N*32
    float* buf_y = buf_x + (size_t)B_ * N_ * 32;      // B*N*32

    const int T = 256;
    int gBN  = (B_ * N_ + T - 1) / T;
    int gBE  = (B_ * E_ + T - 1) / T;

    // degree normalization (shared across all three layers)
    k_init_deg <<<gBN, T, 0, stream>>>(dinv);
    k_count_deg<<<gBE, T, 0, stream>>>(ei, dinv);
    k_rsqrt_deg<<<gBN, T, 0, stream>>>(dinv);

    // layer 1: 2 -> 16
    k_linear<2, 16> <<<gBN, T, 0, stream>>>(nodes, W1, buf_h);
    k_selfinit<16>  <<<gBN, T, 0, stream>>>(buf_h, dinv, buf_x);
    k_scatter<16>   <<<gBE, T, 0, stream>>>(buf_h, dinv, ei, buf_x);
    k_biasrelu<16>  <<<(B_ * N_ * 16 + T - 1) / T, T, 0, stream>>>(buf_x, b1);

    // layer 2: 16 -> 32
    k_linear<16, 32><<<gBN, T, 0, stream>>>(buf_x, W2, buf_h);
    k_selfinit<32>  <<<gBN, T, 0, stream>>>(buf_h, dinv, buf_y);
    k_scatter<32>   <<<gBE, T, 0, stream>>>(buf_h, dinv, ei, buf_y);
    k_biasrelu<32>  <<<(B_ * N_ * 32 + T - 1) / T, T, 0, stream>>>(buf_y, b2);

    // layer 3: 32 -> 32
    k_linear<32, 32><<<gBN, T, 0, stream>>>(buf_y, W3, buf_h);
    k_selfinit<32>  <<<gBN, T, 0, stream>>>(buf_h, dinv, buf_x);
    k_scatter<32>   <<<gBE, T, 0, stream>>>(buf_h, dinv, ei, buf_x);
    k_biasrelu<32>  <<<(B_ * N_ * 32 + T - 1) / T, T, 0, stream>>>(buf_x, b3);

    // fused gumbel-softmax + WMMA pooling
    k_zero_out<<<(out_size + T - 1) / T, T, 0, stream>>>(out, out_size);
    k_pool<<<B_ * SPLIT, 32, 0, stream>>>(buf_x, logits, gu, out);
}
